// TTTLayer_60395830116900
// MI455X (gfx1250) — matrix operand order repaired and checked by reading, not compiled, not run on hardware
//
#include <hip/hip_runtime.h>
#include <hip/hip_bf16.h>

typedef __bf16 bf16_t;
typedef __attribute__((ext_vector_type(16))) __bf16 v16bf;
typedef __attribute__((ext_vector_type(8)))  __bf16 v8bf;
typedef __attribute__((ext_vector_type(8)))  float  v8f;

#define HID 768
#define NH 12
#define DH 64
#define SEQ 8192
#define NB 4
#define KMB 16
#define NMB (SEQ / KMB)
#define MROWS (NB * SEQ)
#define LN_EPS 1e-5f

__device__ __forceinline__ v8f wmma_bf16(v16bf a, v16bf b, v8f c) {
  return __builtin_amdgcn_wmma_f32_16x16x32_bf16(false, a, false, b, (short)0, c, false, false);
}

__device__ __forceinline__ float redrow16(float v) {
  v += __shfl_xor(v, 1, 32);
  v += __shfl_xor(v, 2, 32);
  v += __shfl_xor(v, 4, 32);
  v += __shfl_xor(v, 8, 32);
  return v;
}

// ---------------- prep kernels ----------------

__global__ void cvt_f32_bf16_kernel(const float* __restrict__ s, bf16_t* __restrict__ d, int n) {
  int i = blockIdx.x * 256 + threadIdx.x;
  if (i < n) d[i] = (bf16_t)s[i];
}

// dst[n*K + k] = src[k*N + n]  (store B transposed so WMMA B-operand rows are contiguous)
__global__ void transpose_bf16_kernel(const float* __restrict__ src, bf16_t* __restrict__ dst,
                                      int K, int N) {
  int i = blockIdx.x * 256 + threadIdx.x;
  if (i < K * N) {
    int n = i % N, k = i / N;
    dst[(size_t)n * K + k] = (bf16_t)src[(size_t)k * N + n];
  }
}

// ---------------- big WMMA GEMM: C(MxN,f32) = A(MxK,bf16) * Bt(NxK,bf16)^T ----------------
// Wave tile: 16 (M) x 64 (N). Block: 8 waves stacked in M -> 128 x 64 tile.
__launch_bounds__(256)
__global__ void gemm_bf16_wmma(const bf16_t* __restrict__ A, const bf16_t* __restrict__ Bt,
                               float* __restrict__ C, int M, int N, int K) {
  const int wave = threadIdx.x >> 5;
  const int lane = threadIdx.x & 31;
  const int l16 = lane & 15;
  const int half = lane >> 4;
  const int m0 = (blockIdx.y * 8 + wave) * 16;
  const int n0 = blockIdx.x * 64;

  const bf16_t* arow = A + (size_t)(m0 + l16) * K;
  v8f zero = {};
  v8f acc[4];
  acc[0] = zero; acc[1] = zero; acc[2] = zero; acc[3] = zero;

  for (int k0 = 0; k0 < K; k0 += 32) {
    v8bf alo = *reinterpret_cast<const v8bf*>(arow + k0 + half * 16);
    v8bf ahi = *reinterpret_cast<const v8bf*>(arow + k0 + half * 16 + 8);
    v16bf a;
#pragma unroll
    for (int e = 0; e < 8; ++e) { a[e] = alo[e]; a[e + 8] = ahi[e]; }
#pragma unroll
    for (int t = 0; t < 4; ++t) {
      const bf16_t* bp = Bt + (size_t)(n0 + t * 16 + l16) * K + k0 + half * 16;
      v16bf b = *reinterpret_cast<const v16bf*>(bp);
      acc[t] = wmma_bf16(a, b, acc[t]);
    }
  }
#pragma unroll
  for (int t = 0; t < 4; ++t)
#pragma unroll
    for (int r = 0; r < 8; ++r)
      C[(size_t)(m0 + half * 8 + r) * N + n0 + t * 16 + l16] = acc[t][r];
}

// ---------------- ttt_lr = sigmoid(x @ lr_w^T + lr_b), stored (B,H,S) ----------------
__global__ void lr_kernel(const float* __restrict__ x, const float* __restrict__ lrw,
                          const float* __restrict__ lrb, float* __restrict__ tl) {
  int t = blockIdx.x * 256 + threadIdx.x;
  if (t >= MROWS * NH) return;
  int h = t % NH;
  int bs = t / NH;
  const float* xr = x + (size_t)bs * HID;
  const float* wr = lrw + (size_t)h * HID;
  float acc = lrb[h];
  for (int d = 0; d < HID; d += 4) {
    acc += xr[d] * wr[d] + xr[d + 1] * wr[d + 1] + xr[d + 2] * wr[d + 2] + xr[d + 3] * wr[d + 3];
  }
  float sig = 1.0f / (1.0f + __expf(-acc));
  int b = bs >> 13, s = bs & (SEQ - 1);
  tl[((size_t)b * NH + h) * SEQ + s] = sig;
}

// ---------------- causal dwconv (taps s-4..s-1) + RoPE ----------------
__global__ void convrope_kernel(const float* __restrict__ xqk,
                                const float* __restrict__ cqw, const float* __restrict__ cqb,
                                const float* __restrict__ ckw, const float* __restrict__ ckb,
                                float* __restrict__ XQ, float* __restrict__ XK) {
  int t = blockIdx.x * 256 + threadIdx.x;
  if (t >= MROWS * NH * (DH / 2)) return;
  int d2 = t & 31;
  int h = (t >> 5) % NH;
  int bs = t / (32 * NH);
  int b = bs >> 13, s = bs & (SEQ - 1);
  int c0 = h * DH + 2 * d2;

  float q0 = cqb[c0], q1 = cqb[c0 + 1], k0 = ckb[c0], k1 = ckb[c0 + 1];
#pragma unroll
  for (int kk = 0; kk < 4; ++kk) {
    int sp = s + kk - 4;
    if (sp >= 0) {
      const float* xr = xqk + ((size_t)(b * SEQ + sp)) * HID + c0;
      q0 += cqw[kk * HID + c0] * xr[0];
      q1 += cqw[kk * HID + c0 + 1] * xr[1];
      k0 += ckw[kk * HID + c0] * xr[0];
      k1 += ckw[kk * HID + c0 + 1] * xr[1];
    }
  }
  float inv = __expf(-((float)(2 * d2) / 64.0f) * 9.210340371976184f); // ln(10000)
  float ang = (float)s * inv;
  float cs = __cosf(ang), sn = __sinf(ang);
  size_t o = ((size_t)(b * SEQ + s)) * HID + c0;
  XQ[o]     = q0 * cs - q1 * sn;
  XQ[o + 1] = q0 * sn + q1 * cs;
  XK[o]     = k0 * cs - k1 * sn;
  XK[o + 1] = k0 * sn + k1 * cs;
}

// ---------------- sequential TTT scan: one wave per (b,h) ----------------
__launch_bounds__(32)
__global__ void ttt_scan_kernel(const float* __restrict__ XQ, const float* __restrict__ XK,
                                const float* __restrict__ XV, const float* __restrict__ TL,
                                const float* __restrict__ ltok,
                                const float* __restrict__ nsc, const float* __restrict__ nbi,
                                const float* __restrict__ W1g, const float* __restrict__ b1g,
                                float* __restrict__ Z) {
  __shared__ float sW1T[64 * 64];   // sW1T[e_out*64 + d_in]
  __shared__ float sb1[64];
  __shared__ float sK[16 * 64];     // sK[j*64 + d]
  __shared__ float sKT[64 * 16];    // sKT[d*16 + j]
  __shared__ float sGT[64 * 16];    // grad transposed: sGT[n*16 + m]
  __shared__ float sAttn[16 * 16];
  __shared__ float sEta[16];
  __shared__ float sTok[16];

  const int bh = blockIdx.x;
  const int b = bh / NH, h = bh % NH;
  const int lane = threadIdx.x & 31;
  const int l16 = lane & 15;
  const int half = lane >> 4;

  for (int i = lane; i < 64 * 64; i += 32) {
    int d = i >> 6, e = i & 63;
    sW1T[e * 64 + d] = W1g[(size_t)h * 4096 + d * 64 + e];
  }
  for (int i = lane; i < 64; i += 32) sb1[i] = b1g[h * 64 + i];
  if (lane < 16) sTok[lane] = fmaxf(1.0f / (float)(lane + 1) + ltok[lane], 0.0f);
  __syncthreads();
  const float tok15 = sTok[15];

  float gs[4], gb[4];
#pragma unroll
  for (int t = 0; t < 4; ++t) {
    int n = t * 16 + l16;
    gs[t] = nsc[h * 64 + n];
    gb[t] = nbi[h * 64 + n];
  }
  const size_t baseBS = (size_t)b * SEQ;
  const v8f zero = {};

  for (int mb = 0; mb < NMB; ++mb) {
    const int s0 = mb * KMB;
    // stage K tile (row-major + transposed), eta row
    {
      const float* kp = XK + (baseBS + s0 + l16) * HID + h * 64 + half * 32;
#pragma unroll
      for (int i = 0; i < 32; ++i) {
        float v = kp[i];
        int d = half * 32 + i;
        sK[l16 * 64 + d] = v;
        sKT[d * 16 + l16] = v;
      }
      if (lane < 16) sEta[lane] = (1.0f / 64.0f) * TL[((size_t)b * NH + h) * SEQ + s0 + lane];
    }
    if (mb + 1 < NMB) { // prefetch next tiles (global_prefetch_b8)
      __builtin_prefetch(XK + (baseBS + s0 + 16 + l16) * HID + h * 64 + half * 32, 0, 1);
      __builtin_prefetch(XQ + (baseBS + s0 + 16 + l16) * HID + h * 64 + half * 32, 0, 1);
    }
    __syncthreads();

    // A fragments: k (from LDS), q (from global)
    v16bf ak[2], aq[2];
#pragma unroll
    for (int ks = 0; ks < 2; ++ks) {
      const float* qp = XQ + (baseBS + s0 + l16) * HID + h * 64 + ks * 32;
#pragma unroll
      for (int e = 0; e < 8; ++e) {
        ak[ks][e]     = (bf16_t)sK[l16 * 64 + ks * 32 + half * 16 + e];
        ak[ks][e + 8] = (bf16_t)sK[l16 * 64 + ks * 32 + half * 16 + 8 + e];
        aq[ks][e]     = (bf16_t)qp[half * 16 + e];
        aq[ks][e + 8] = (bf16_t)qp[half * 16 + 8 + e];
      }
    }

    // Z1 = k@W1, Zq = q@W1 (4 n-tiles x 2 k-steps each)
    v8f z1[4], zb[4];
#pragma unroll
    for (int t = 0; t < 4; ++t) { z1[t] = zero; zb[t] = zero; }
#pragma unroll
    for (int ks = 0; ks < 2; ++ks) {
#pragma unroll
      for (int t = 0; t < 4; ++t) {
        const float* wp = &sW1T[(t * 16 + l16) * 64 + ks * 32 + half * 16];
        v16bf bw;
#pragma unroll
        for (int e = 0; e < 16; ++e) bw[e] = (bf16_t)wp[e];
        z1[t] = wmma_bf16(ak[ks], bw, z1[t]);
        zb[t] = wmma_bf16(aq[ks], bw, zb[t]);
      }
    }
#pragma unroll
    for (int t = 0; t < 4; ++t) {
      const float bias = sb1[t * 16 + l16];
#pragma unroll
      for (int r = 0; r < 8; ++r) z1[t][r] += bias;
    }

    // Attn = q k^T  (16x16, masked later)
    v8f attn = zero;
#pragma unroll
    for (int ks = 0; ks < 2; ++ks) {
      const float* kp2 = &sK[l16 * 64 + ks * 32 + half * 16];
      v16bf bk;
#pragma unroll
      for (int e = 0; e < 16; ++e) bk[e] = (bf16_t)kp2[e];
      attn = wmma_bf16(aq[ks], bk, attn);
    }
#pragma unroll
    for (int r = 0; r < 8; ++r) sAttn[(half * 8 + r) * 16 + l16] = attn[r];

    // ln_l2_bwd(Z1, v-k) -> grad -> sGT (transposed)
    float mu1[8], rs1[8], mgx[8], mgxh[8];
#pragma unroll
    for (int r = 0; r < 8; ++r) {
      float s = z1[0][r] + z1[1][r] + z1[2][r] + z1[3][r];
      float s2 = z1[0][r] * z1[0][r] + z1[1][r] * z1[1][r] + z1[2][r] * z1[2][r] + z1[3][r] * z1[3][r];
      s = redrow16(s);
      s2 = redrow16(s2);
      float mu = s * (1.0f / 64.0f);
      float var = s2 * (1.0f / 64.0f) - mu * mu;
      mu1[r] = mu;
      rs1[r] = rsqrtf(var + LN_EPS);
    }
#pragma unroll
    for (int r = 0; r < 8; ++r) {
      const int m = half * 8 + r;
      float pg = 0.f, ph = 0.f;
#pragma unroll
      for (int t = 0; t < 4; ++t) {
        const int n = t * 16 + l16;
        float xh = (z1[t][r] - mu1[r]) * rs1[r];
        float tgt = XV[(baseBS + s0 + m) * HID + h * 64 + n] - sK[m * 64 + n];
        float gx = (xh * gs[t] + gb[t] - tgt) * gs[t];
        pg += gx;
        ph += gx * xh;
      }
      mgx[r] = redrow16(pg) * (1.0f / 64.0f);
      mgxh[r] = redrow16(ph) * (1.0f / 64.0f);
    }
#pragma unroll
    for (int r = 0; r < 8; ++r) {
      const int m = half * 8 + r;
#pragma unroll
      for (int t = 0; t < 4; ++t) {
        const int n = t * 16 + l16;
        float xh = (z1[t][r] - mu1[r]) * rs1[r];
        float tgt = XV[(baseBS + s0 + m) * HID + h * 64 + n] - sK[m * 64 + n];
        float gx = (xh * gs[t] + gb[t] - tgt) * gs[t];
        sGT[n * 16 + m] = (gx - mgx[r] - xh * mgxh[r]) * rs1[r];
      }
    }
    __syncthreads();

    // Z1_bar = q@W1 + b1 - [e .* (Attn + tril)] @ grad   (A built negated; K=16 padded to 32)
    v16bf acm;
#pragma unroll
    for (int e = 0; e < 8; ++e) {
      const int kk = half * 8 + e;
      float val = (kk <= l16) ? (-(sTok[l16] * sEta[kk] * (sAttn[l16 * 16 + kk] + 1.0f))) : 0.0f;
      acm[e] = (bf16_t)val;
      acm[e + 8] = (bf16_t)0.0f;
    }
#pragma unroll
    for (int t = 0; t < 4; ++t) {
      const float* gp = &sGT[(t * 16 + l16) * 16];
      v16bf bg;
#pragma unroll
      for (int e = 0; e < 16; ++e) bg[e] = (bf16_t)((e < 8) ? gp[half * 8 + e] : 0.0f);
      zb[t] = wmma_bf16(acm, bg, zb[t]);
    }
#pragma unroll
    for (int t = 0; t < 4; ++t) {
      const float bias = sb1[t * 16 + l16];
#pragma unroll
      for (int r = 0; r < 8; ++r) zb[t][r] += bias;
    }

    // out = q + ln_fwd(Z1_bar)
    float mu2[8], rs2[8];
#pragma unroll
    for (int r = 0; r < 8; ++r) {
      float s = zb[0][r] + zb[1][r] + zb[2][r] + zb[3][r];
      float s2 = zb[0][r] * zb[0][r] + zb[1][r] * zb[1][r] + zb[2][r] * zb[2][r] + zb[3][r] * zb[3][r];
      s = redrow16(s);
      s2 = redrow16(s2);
      float mu = s * (1.0f / 64.0f);
      float var = s2 * (1.0f / 64.0f) - mu * mu;
      mu2[r] = mu;
      rs2[r] = rsqrtf(var + LN_EPS);
    }
#pragma unroll
    for (int r = 0; r < 8; ++r) {
      const int m = half * 8 + r;
      const size_t rowoff = (baseBS + s0 + m) * HID + h * 64;
#pragma unroll
      for (int t = 0; t < 4; ++t) {
        const int n = t * 16 + l16;
        float xh = (zb[t][r] - mu2[r]) * rs2[r];
        Z[rowoff + n] = XQ[rowoff + n] + xh * gs[t] + gb[t];
      }
    }

    // b1 <- b1 - tok15 * sum_j eta_j * grad[j][:]
#pragma unroll
    for (int c = 0; c < 2; ++c) {
      const int d = lane * 2 + c;
      float s = 0.f;
#pragma unroll
      for (int j = 0; j < 16; ++j) s += sEta[j] * sGT[d * 16 + j];
      sb1[d] -= tok15 * s;
    }

    // W1 <- W1 - (last_e .* k)^T @ grad   (A built negated; 4x4 tiles, K=16 padded)
#pragma unroll
    for (int mt = 0; mt < 4; ++mt) {
      const int drow = mt * 16 + l16;
      v16bf aw;
#pragma unroll
      for (int e = 0; e < 8; ++e) {
        const int j = half * 8 + e;
        aw[e] = (bf16_t)(-(tok15 * sEta[j] * sKT[drow * 16 + j]));
        aw[e + 8] = (bf16_t)0.0f;
      }
#pragma unroll
      for (int t = 0; t < 4; ++t) {
        const float* gp = &sGT[(t * 16 + l16) * 16];
        v16bf bg;
#pragma unroll
        for (int e = 0; e < 16; ++e) bg[e] = (bf16_t)((e < 8) ? gp[half * 8 + e] : 0.0f);
        v8f cw;
        const int col = t * 16 + l16;
        const int row0 = mt * 16 + half * 8;
#pragma unroll
        for (int r = 0; r < 8; ++r) cw[r] = sW1T[col * 64 + row0 + r];
        cw = wmma_bf16(aw, bg, cw);
#pragma unroll
        for (int r = 0; r < 8; ++r) sW1T[col * 64 + row0 + r] = cw[r];
      }
    }
    __syncthreads();
  }
}

// ---------------- post-norm * gelu(gate) -> bf16 rows ----------------
__global__ void postgate_kernel(const float* __restrict__ Zb, const float* __restrict__ graw,
                                const float* __restrict__ psc, const float* __restrict__ pbi,
                                bf16_t* __restrict__ ybf) {
  const int row = blockIdx.x;
  const int tid = threadIdx.x;
  const float* zr = Zb + (size_t)row * HID;
  float v0 = zr[tid], v1 = zr[tid + 256], v2 = zr[tid + 512];
  float s = v0 + v1 + v2;
  float s2 = v0 * v0 + v1 * v1 + v2 * v2;
#pragma unroll
  for (int m = 1; m < 32; m <<= 1) {
    s += __shfl_xor(s, m, 32);
    s2 += __shfl_xor(s2, m, 32);
  }
  __shared__ float red[16];
  int w = tid >> 5, ln = tid & 31;
  if (ln == 0) { red[w] = s; red[w + 8] = s2; }
  __syncthreads();
  if (tid == 0) {
    float a = 0.f, b2 = 0.f;
    for (int i = 0; i < 8; ++i) { a += red[i]; b2 += red[i + 8]; }
    red[0] = a; red[8] = b2;
  }
  __syncthreads();
  float mu = red[0] * (1.0f / 768.0f);
  float var = red[8] * (1.0f / 768.0f) - mu * mu;
  float rs = rsqrtf(var + LN_EPS);
  const float* gr = graw + (size_t)row * HID;
#pragma unroll
  for (int kb = 0; kb < 3; ++kb) {
    int c = tid + kb * 256;
    float zn = (zr[c] - mu) * rs * psc[c] + pbi[c];
    float g = gr[c];
    float gg = 0.5f * g * (1.0f + tanhf(0.7978845608028654f * (g + 0.044715f * g * g * g)));
    ybf[(size_t)row * HID + c] = (bf16_t)(zn * gg);
  }
}

// ---------------- launch ----------------
extern "C" void kernel_launch(void* const* d_in, const int* in_sizes, int n_in,
                              void* d_out, int out_size, void* d_ws, size_t ws_size,
                              hipStream_t stream) {
  (void)in_sizes; (void)n_in; (void)out_size; (void)ws_size;
  const float* x    = (const float*)d_in[0];
  const float* wq   = (const float*)d_in[1];
  const float* wv   = (const float*)d_in[2];
  const float* wo   = (const float*)d_in[3];
  const float* wg   = (const float*)d_in[4];
  const float* cqw  = (const float*)d_in[5];
  const float* cqb  = (const float*)d_in[6];
  const float* ckw  = (const float*)d_in[7];
  const float* ckb  = (const float*)d_in[8];
  const float* lrw  = (const float*)d_in[9];
  const float* lrb  = (const float*)d_in[10];
  const float* ltok = (const float*)d_in[11];
  const float* nsc  = (const float*)d_in[12];
  const float* nbi  = (const float*)d_in[13];
  const float* psc  = (const float*)d_in[14];
  const float* pbi  = (const float*)d_in[15];
  const float* W1   = (const float*)d_in[16];
  const float* b1   = (const float*)d_in[17];
  float* out = (float*)d_out;

  char* ws = (char*)d_ws;
  const size_t XBF_B  = (size_t)MROWS * HID * 2;   // 50331648
  const size_t WT_B   = (size_t)HID * HID * 2;     // 1179648
  const size_t F32_B  = (size_t)MROWS * HID * 4;   // 100663296
  size_t off = 0;
  bf16_t* xbf = (bf16_t*)(ws + off); off += XBF_B;
  bf16_t* wqt = (bf16_t*)(ws + off); off += WT_B;
  bf16_t* wvt = (bf16_t*)(ws + off); off += WT_B;
  bf16_t* wgt = (bf16_t*)(ws + off); off += WT_B;
  bf16_t* wot = (bf16_t*)(ws + off); off += WT_B;
  float* xqk  = (float*)(ws + off); off += F32_B;
  float* xv   = (float*)(ws + off); off += F32_B;
  float* graw = (float*)(ws + off); off += F32_B;
  float* XQ   = (float*)(ws + off); off += F32_B;
  float* XK   = (float*)(ws + off); off += F32_B;
  float* tl   = (float*)(ws + off); off += (size_t)NB * NH * SEQ * 4;
  float* Zbuf = xqk;      // xqk dead after convrope
  bf16_t* ybf = xbf;      // xbf dead after the three input GEMMs

  const int nX = MROWS * HID;
  cvt_f32_bf16_kernel<<<(nX + 255) / 256, 256, 0, stream>>>(x, xbf, nX);
  const int nW = HID * HID;
  transpose_bf16_kernel<<<(nW + 255) / 256, 256, 0, stream>>>(wq, wqt, HID, HID);
  transpose_bf16_kernel<<<(nW + 255) / 256, 256, 0, stream>>>(wv, wvt, HID, HID);
  transpose_bf16_kernel<<<(nW + 255) / 256, 256, 0, stream>>>(wg, wgt, HID, HID);
  transpose_bf16_kernel<<<(nW + 255) / 256, 256, 0, stream>>>(wo, wot, HID, HID);

  dim3 gg(HID / 64, MROWS / 128);
  gemm_bf16_wmma<<<gg, 256, 0, stream>>>(xbf, wqt, xqk, MROWS, HID, HID);
  gemm_bf16_wmma<<<gg, 256, 0, stream>>>(xbf, wvt, xv, MROWS, HID, HID);
  gemm_bf16_wmma<<<gg, 256, 0, stream>>>(xbf, wgt, graw, MROWS, HID, HID);

  const int nLR = MROWS * NH;
  lr_kernel<<<(nLR + 255) / 256, 256, 0, stream>>>(x, lrw, lrb, tl);

  const int nCR = MROWS * NH * (DH / 2);
  convrope_kernel<<<(nCR + 255) / 256, 256, 0, stream>>>(xqk, cqw, cqb, ckw, ckb, XQ, XK);

  ttt_scan_kernel<<<NB * NH, 32, 0, stream>>>(XQ, XK, xv, tl, ltok, nsc, nbi, W1, b1, Zbuf);

  postgate_kernel<<<MROWS, 256, 0, stream>>>(Zbuf, graw, psc, pbi, ybf);

  gemm_bf16_wmma<<<gg, 256, 0, stream>>>(ybf, wot, out, MROWS, HID, HID);
}
